// NodeAttentionLayer_50878182589041
// MI455X (gfx1250) — matrix operand order repaired and checked by reading, not compiled, not run on hardware
//
#include <hip/hip_runtime.h>
#include <hip/hip_bf16.h>

typedef __attribute__((ext_vector_type(16))) _Float16 v16h;
typedef __attribute__((ext_vector_type(8)))  _Float16 h8;
typedef __attribute__((ext_vector_type(4)))  _Float16 h4;
typedef __attribute__((ext_vector_type(8)))  float    v8f;
typedef __attribute__((ext_vector_type(4)))  unsigned int u32x4;
typedef __attribute__((ext_vector_type(8)))  int      i32x8;
typedef __attribute__((ext_vector_type(4)))  int      i32x4;

#define B_   128
#define N_   512
#define FIN  128
#define FOUT 64
#define AP   136   // LDS pitch (halves) for kernel1 tiles
#define HP   520   // LDS pitch (halves/floats) for kernel2 slabs (512 + 8 pad)

// ---- DPP-based wave reductions (no LDS traffic, no bpermute clamping) ----
// xor masks {1,2,7,15} span the 16-lane space: quad_perm(0xB1), quad_perm(0x4E),
// row_half_mirror(0x141), row_mirror(0x140).
template <int CTRL>
static __device__ __forceinline__ float dpp_bfly(float x) {
  return __int_as_float(__builtin_amdgcn_update_dpp(
      0, __float_as_int(x), CTRL, 0xF, 0xF, true));
}
static __device__ __forceinline__ float red_add16(float x) {
  x += dpp_bfly<0x0B1>(x);
  x += dpp_bfly<0x04E>(x);
  x += dpp_bfly<0x141>(x);
  x += dpp_bfly<0x140>(x);
  return x;                       // all lanes hold their 16-lane row sum
}
static __device__ __forceinline__ float red_add32(float x) {
  x = red_add16(x);
  int y = __builtin_amdgcn_permlanex16(__float_as_int(x), __float_as_int(x),
                                       0x76543210, 0xfedcba98, true, false);
  return x + __int_as_float(y);   // + other 16-lane row
}
static __device__ __forceinline__ float red_max32(float x) {
  x = fmaxf(x, dpp_bfly<0x0B1>(x));
  x = fmaxf(x, dpp_bfly<0x04E>(x));
  x = fmaxf(x, dpp_bfly<0x141>(x));
  x = fmaxf(x, dpp_bfly<0x140>(x));
  int y = __builtin_amdgcn_permlanex16(__float_as_int(x), __float_as_int(x),
                                       0x76543210, 0xfedcba98, true, false);
  return fmaxf(x, __int_as_float(y));
}

// Assemble a WMMA 16-bit operand from two 16-byte LDS chunks
// (per-lane A and B layout: K {sel..sel+7} then {sel+16..sel+23})
static __device__ __forceinline__ v16h ld_frag(const _Float16* p0, const _Float16* p1) {
  h8 lo = *(const h8*)p0;
  h8 hi = *(const h8*)p1;
  return __builtin_shufflevector(lo, hi, 0,1,2,3,4,5,6,7,8,9,10,11,12,13,14,15);
}

// ---------------------------------------------------------------------------
// Kernel 1: h = X @ W (f16 WMMA, f32 acc), fused s = h.a_src, t = h.a_dst
// block = 256 thr (8 waves, 2x4 tiles of 16x16), 32 rows per block.
// h is written TRANSPOSED per batch: hbufT[(b*FOUT + n)*N_ + j], so each lane
// stores its 8 consecutive-j halves as one global_store_b128.
// ---------------------------------------------------------------------------
__global__ __launch_bounds__(256)
void gat_h_kernel(const float* __restrict__ X, const float* __restrict__ W,
                  const float* __restrict__ avec,
                  _Float16* __restrict__ hbufT, float* __restrict__ sbuf,
                  float* __restrict__ tbuf) {
  __shared__ _Float16 Xs[32 * AP];
  __shared__ _Float16 WT[FOUT * AP];
  __shared__ float sred[32];
  __shared__ float tred[32];

  const int tid = threadIdx.x;
  const int rowBase = blockIdx.x * 32;          // flat (b*N + j) row base
  const int b  = rowBase >> 9;                  // N_ = 512
  const int j0 = rowBase & (N_ - 1);

  // Stage X tile: float4 global load -> 4 halves (ds_store_b64)
  for (int idx = tid; idx < 32 * FIN / 4; idx += 256) {
    int r = idx >> 5, kc = (idx & 31) << 2;
    float4 f = *(const float4*)&X[(size_t)(rowBase + r) * FIN + kc];
    h4 hv = { (_Float16)f.x, (_Float16)f.y, (_Float16)f.z, (_Float16)f.w };
    *(h4*)&Xs[r * AP + kc] = hv;
  }
  // Stage W transposed: WT[n][k]; float4 read over n, scalar b16 LDS stores
  for (int idx = tid; idx < FIN * FOUT / 4; idx += 256) {
    int k = idx >> 4, n4 = (idx & 15) << 2;
    float4 f = *(const float4*)&W[k * FOUT + n4];
    WT[(n4 + 0) * AP + k] = (_Float16)f.x;
    WT[(n4 + 1) * AP + k] = (_Float16)f.y;
    WT[(n4 + 2) * AP + k] = (_Float16)f.z;
    WT[(n4 + 3) * AP + k] = (_Float16)f.w;
  }
  if (tid < 32) { sred[tid] = 0.f; tred[tid] = 0.f; }
  __syncthreads();

  const int lane = tid & 31, w = tid >> 5;
  const int mt = w >> 2, nt = w & 3;
  const int l16   = lane & 15;
  const int hi    = lane >> 4;
  const int sel8  = hi << 3;   // A: +8 K for upper half-wave
  const int off16 = hi << 4;   // unused; B also uses sel8 (+8 K)
  const _Float16* arow = &Xs[(mt * 16 + l16) * AP];
  const _Float16* brow = &WT[(nt * 16 + l16) * AP];

  v8f acc = {};
#pragma unroll
  for (int k0 = 0; k0 < FIN; k0 += 32) {
    v16h a  = ld_frag(arow + k0 + sel8,  arow + k0 + sel8 + 16);
    v16h bm = ld_frag(brow + k0 + sel8, brow + k0 + sel8 + 16);
    acc = __builtin_amdgcn_wmma_f32_16x16x32_f16(false, a, false, bm,
                                                 (short)0, acc, false, false);
  }

  const int col = nt * 16 + l16;
  const float asrc = avec[col];
  const float adst = avec[FOUT + col];

  // Transposed store: rows j0+mt*16+8*hi+v are consecutive in v -> one b128.
  // s/t partials: DPP butterfly over the 16 columns of each half-wave, then a
  // single ds_add_f32 per row per wave (2 active lanes).
  h8 hpack;
#pragma unroll
  for (int v = 0; v < 8; ++v) {
    int lr = mt * 16 + (hi << 3) + v;
    float hv = acc[v];
    hpack[v] = (_Float16)hv;
    float sv = red_add16(hv * asrc);
    float tv = red_add16(hv * adst);
    if (l16 == 0) {
      atomicAdd(&sred[lr], sv);   // ds_add_f32 (4 waves share each row)
      atomicAdd(&tred[lr], tv);
    }
  }
  {
    int jrow = j0 + mt * 16 + (hi << 3);
    *(h8*)&hbufT[((size_t)b * FOUT + col) * N_ + jrow] = hpack;
  }
  __syncthreads();
  if (tid < 32) {
    sbuf[rowBase + tid] = sred[tid];
    tbuf[rowBase + tid] = tred[tid];
  }
}

// ---------------------------------------------------------------------------
// Kernel 2: fused masked-softmax attention + h' = att @ h + leaky_relu(0.01)
// block = (batch b, 32-row tile). LDS: h^T[64][HP] f16, att[32][HP] f16,
// logits/probs esc[32][HP] f32, t[512] f32, s[32] f32  (~165 KB dynamic LDS)
// h^T slab is staged by the Tensor Data Mover (one descriptor, LDS padding
// 4 DW per 256 DW giving the conflict-free pitch HP).
// ---------------------------------------------------------------------------
__global__ __launch_bounds__(256)
void gat_attn_kernel(const int* __restrict__ adj, const _Float16* __restrict__ hbufT,
                     const float* __restrict__ sbuf, const float* __restrict__ tbuf,
                     float* __restrict__ out) {
  extern __shared__ char smem[];
  _Float16* hT  = (_Float16*)smem;                 // FOUT * HP
  _Float16* att = hT + FOUT * HP;                  // 32 * HP
  float*    esc = (float*)(att + 32 * HP);         // 32 * HP
  float*    tt  = esc + 32 * HP;                   // N_
  float*    ss  = tt + N_;                         // 32

  const int tid = threadIdx.x;
  const int b = blockIdx.x >> 4;
  const int rowBase = (blockIdx.x & 15) * 32;

  const _Float16* hbT = hbufT + (size_t)b * FOUT * N_;

#if defined(__has_builtin) && __has_builtin(__builtin_amdgcn_tensor_load_to_lds)
  // ---- TDM: one descriptor moves the whole 64x512 f16 tile into LDS ----
  if ((tid >> 5) == 0) {                           // one wave issues the DMA
    unsigned lds_base = (unsigned)(uintptr_t)(void*)hT;
    unsigned long long ga = (unsigned long long)(uintptr_t)hbT;
    u32x4 g0;
    g0[0] = 1u;                                    // count=1 (valid user D#)
    g0[1] = lds_base;                              // lds_addr
    g0[2] = (unsigned)ga;                          // global_addr[31:0]
    g0[3] = (unsigned)(ga >> 32) | (2u << 30);     // global_addr[56:32] | type=2
    i32x8 g1;
    g1[0] = (1 << 16)      |                       // data_size = 2 bytes
            (1 << 20)      |                       // pad_enable
            (7 << 22)      |                       // pad_interval = 256 DWORDs
            (3 << 25);                             // pad_amount   = 4 DWORDs
    g1[1] = (int)(512u << 16);                     // tensor_dim0 = 512 (lo16)
    g1[2] = (int)(64u << 16);                      // tensor_dim1 = 64  (lo16)
    g1[3] = (int)(512u << 16);                     // tile_dim0 = 512
    g1[4] = 64;                                    // tile_dim1 = 64, tile_dim2 = 0
    g1[5] = 512;                                   // tensor_dim0_stride = 512
    g1[6] = 0;
    g1[7] = 0;
    i32x4 z4 = { 0, 0, 0, 0 };
    i32x8 z8 = { 0, 0, 0, 0, 0, 0, 0, 0 };
    __builtin_amdgcn_tensor_load_to_lds(g0, g1, z4, z4, z8, 0);  // 6-arg form
    __builtin_amdgcn_s_wait_tensorcnt((short)0);   // s_wait_tensorcnt 0
  }
#else
  for (int idx = tid; idx < FOUT * N_ / 8; idx += 256) {
    int n = idx >> 6, jc = (idx & 63) << 3;
    *(h8*)&hT[n * HP + jc] = *(const h8*)&hbT[n * N_ + jc];
  }
#endif
  for (int j = tid; j < N_; j += 256) tt[j] = tbuf[b * N_ + j];
  if (tid < 32) ss[tid] = sbuf[b * N_ + rowBase + tid];
  __syncthreads();

  const int lane = tid & 31, w = tid >> 5;

  // ---- masked softmax: each wave owns 4 rows; adj read as int4 ----
  const int* adjb = adj + ((size_t)b * N_ + rowBase) * N_;
  for (int rr = 0; rr < 4; ++rr) {
    const int i = w * 4 + rr;
    const float si = ss[i];
    const int4* adjrow4 = (const int4*)(adjb + (size_t)i * N_);
    float m = -3.0e38f;
#pragma unroll
    for (int it = 0; it < N_ / 128; ++it) {        // 4 iters, 4 j per lane
      int j4 = lane + (it << 5);
      __builtin_prefetch((const int*)(adjrow4 + j4) + 2 * N_, 0, 0);
      int4 av = adjrow4[j4];
      float4 tv = *(const float4*)&tt[j4 << 2];
      float4 x;
      x.x = si + tv.x; x.x = x.x > 0.f ? x.x : 0.2f * x.x; x.x = av.x > 0 ? x.x : -9.0e15f;
      x.y = si + tv.y; x.y = x.y > 0.f ? x.y : 0.2f * x.y; x.y = av.y > 0 ? x.y : -9.0e15f;
      x.z = si + tv.z; x.z = x.z > 0.f ? x.z : 0.2f * x.z; x.z = av.z > 0 ? x.z : -9.0e15f;
      x.w = si + tv.w; x.w = x.w > 0.f ? x.w : 0.2f * x.w; x.w = av.w > 0 ? x.w : -9.0e15f;
      *(float4*)&esc[i * HP + (j4 << 2)] = x;
      m = fmaxf(m, fmaxf(fmaxf(x.x, x.y), fmaxf(x.z, x.w)));
    }
    m = red_max32(m);                              // DPP + permlanex16

    float sum = 0.f;
#pragma unroll
    for (int it = 0; it < N_ / 128; ++it) {        // exp once, keep p in esc
      int j4 = lane + (it << 5);
      float4 x = *(const float4*)&esc[i * HP + (j4 << 2)];
      x.x = __expf(x.x - m); x.y = __expf(x.y - m);
      x.z = __expf(x.z - m); x.w = __expf(x.w - m);
      *(float4*)&esc[i * HP + (j4 << 2)] = x;
      sum += (x.x + x.y) + (x.z + x.w);
    }
    sum = red_add32(sum);                          // DPP + permlanex16
    const float rinv = 1.0f / sum;
#pragma unroll
    for (int it = 0; it < N_ / 128; ++it) {        // normalize -> f16 probs
      int j4 = lane + (it << 5);
      float4 x = *(const float4*)&esc[i * HP + (j4 << 2)];
      h4 p = { (_Float16)(x.x * rinv), (_Float16)(x.y * rinv),
               (_Float16)(x.z * rinv), (_Float16)(x.w * rinv) };
      *(h4*)&att[i * HP + (j4 << 2)] = p;
    }
  }
  __syncthreads();

  // ---- h' tile = att(32x512) @ h(512x64) via f16 WMMA ----
  const int mt = w >> 2, nt = w & 3;
  const int l16   = lane & 15;
  const int hi    = lane >> 4;
  const int sel8  = hi << 3;
  const int off16 = hi << 4;
  const _Float16* arow = &att[(mt * 16 + l16) * HP];
  const _Float16* brow = &hT[(nt * 16 + l16) * HP];

  v8f acc = {};
#pragma unroll 4
  for (int k0 = 0; k0 < N_; k0 += 32) {
    v16h a  = ld_frag(arow + k0 + sel8,  arow + k0 + sel8 + 16);
    v16h bm = ld_frag(brow + k0 + sel8, brow + k0 + sel8 + 16);
    acc = __builtin_amdgcn_wmma_f32_16x16x32_f16(false, a, false, bm,
                                                 (short)0, acc, false, false);
  }

  const int col = nt * 16 + l16;
#pragma unroll
  for (int v = 0; v < 8; ++v) {
    int lr = mt * 16 + (hi << 3) + v;
    float x = acc[v];
    x = x > 0.f ? x : 0.01f * x;                   // final leaky_relu
    out[((size_t)b * N_ + rowBase + lr) * FOUT + col] = x;
  }
}

// ---------------------------------------------------------------------------
extern "C" void kernel_launch(void* const* d_in, const int* in_sizes, int n_in,
                              void* d_out, int out_size, void* d_ws, size_t ws_size,
                              hipStream_t stream) {
  const float* X    = (const float*)d_in[0];   // input_hid (B,N,FIN) f32
  const int*   adj  = (const int*)d_in[1];     // adj (B,N,N) i32
  const float* W    = (const float*)d_in[2];   // W (FIN,FOUT) f32
  const float* avec = (const float*)d_in[3];   // a (2*FOUT,1) f32
  float* out = (float*)d_out;

  // Workspace: hT (f16, per-batch transposed), s, t
  _Float16* hbufT = (_Float16*)d_ws;
  float* sbuf = (float*)((char*)d_ws + (size_t)B_ * N_ * FOUT * sizeof(_Float16));
  float* tbuf = sbuf + B_ * N_;

  const size_t smem = (size_t)(FOUT * HP + 32 * HP) * sizeof(_Float16) +
                      (size_t)(32 * HP + N_ + 32) * sizeof(float);   // ~165 KB
  (void)hipFuncSetAttribute((const void*)gat_attn_kernel,
                            hipFuncAttributeMaxDynamicSharedMemorySize, (int)smem);

  gat_h_kernel<<<(B_ * N_) / 32, 256, 0, stream>>>(X, W, avec, hbufT, sbuf, tbuf);
  gat_attn_kernel<<<B_ * 16, 256, smem, stream>>>(adj, hbufT, sbuf, tbuf, out);
}
